// SPD_72756745994677
// MI455X (gfx1250) — matrix operand order repaired and checked by reading, not compiled, not run on hardware
//
#include <hip/hip_runtime.h>
#include <math.h>

// ---------------------------------------------------------------------------
// CDNA5 / gfx1250 SPD forward. All 1x1 convs -> bf16 WMMA GEMMs (f32 accum),
// LDS-staged tiles with pre-swizzled WMMA fragments.
// ---------------------------------------------------------------------------

typedef __attribute__((ext_vector_type(16))) __bf16 v16bf;
typedef __attribute__((ext_vector_type(2)))  __bf16 v2bf;
typedef __attribute__((ext_vector_type(8)))  float  v8f;

#define BN_INV 0.9999950000374996f   // 1/sqrt(1+1e-5)
#define BATCH 8
#define NPTS  2048
#define NPTS2 4096

__device__ __forceinline__ v8f wmma_bf16(v16bf a, v16bf b, v8f c) {
  return __builtin_amdgcn_wmma_f32_16x16x32_bf16(
      false, a, false, b, (short)0, c, false, false);
}

// A-matrix (16x32, 16-bit) per-lane K offset for VGPR-pair r (this file's contiguous order: k = 16h + 2r)
__device__ __forceinline__ int a_klocal(int r, int h) {
  return (h << 4) + (r << 1);
}

// ---------------------------------------------------------------------------
// Generic multi-source GEMM:  Y[b] = W (OxC) * X[b] (CxN) + epilogue
// Requirements: O % 32 == 0, C % 32 == 0, Nper % 128 == 0,
// concat-source row counts are multiples of 32.
// ---------------------------------------------------------------------------
struct SrcDesc {
  const float* p;  // base
  int rows;        // channels contributed
  int mode;        // 0: X[b][r][n] ; 1: X[b][r] (bcast n) ; 2: X[b][r][n/2]
  int relu;        // relu on input
  int nin;         // leading dim for mode 0/2
};

template <bool UNITW>
__global__ __launch_bounds__(256) void wmma_gemm_kernel(
    const float* __restrict__ W, int w_so, int w_sc, int w_off,
    const float* __restrict__ bias,
    const float* __restrict__ bn_g, const float* __restrict__ bn_b,
    int relu_out, int accum,
    float* __restrict__ Y, int Nout, int ostr, int ooff,
    int O, int C, int Nper,
    SrcDesc s0, SrcDesc s1, SrcDesc s2) {
  // fragment-swizzled LDS: [tile][lane][16 bf16 elements]
  __shared__ __attribute__((aligned(32))) __bf16 sA[2 * 32 * 16];  // 2 KB
  __shared__ __attribute__((aligned(32))) __bf16 sB[8 * 32 * 16];  // 8 KB

  const int t    = threadIdx.x;
  const int lane = t & 31;
  const int wv   = t >> 5;
  const int b    = blockIdx.z;
  const int mB   = blockIdx.y * 32;
  const int nBg  = blockIdx.x * 128;
  const int wvM  = wv >> 2;  // 0..1
  const int wvN  = wv & 3;   // 0..3
  const int h    = lane >> 4;
  const int l16  = lane & 15;

  // per-batch source bases (wave-uniform)
  const float* pb0 = s0.p ? (s0.mode == 1 ? s0.p + (size_t)b * s0.rows
                                          : s0.p + (size_t)b * s0.rows * s0.nin) : nullptr;
  const float* pb1 = s1.p ? (s1.mode == 1 ? s1.p + (size_t)b * s1.rows
                                          : s1.p + (size_t)b * s1.rows * s1.nin) : nullptr;
  const float* pb2 = s2.p ? (s2.mode == 1 ? s2.p + (size_t)b * s2.rows
                                          : s2.p + (size_t)b * s2.rows * s2.nin) : nullptr;

  // loader work assignment
  const int la_m  = t >> 3;         // A: local row 0..31
  const int la_k0 = (t & 7) * 4;    // A: 4 consecutive k
  const int lb_k  = (t >> 4) * 2;   // B: even k (row pair)
  const int lb_c0 = (t & 15) * 8;   // B: 8 consecutive cols

  // precomputed A-store swizzle (h, r constant across ks)
  const int a_mt   = la_m >> 4;
  const int a_lane = ((la_k0 >> 4) << 4) + (la_m & 15);
  const int a_r    = (la_k0 & 15) >> 1;
  // B-store swizzle
  const int b_hB = lb_k >> 4;
  const int b_j  = (lb_k & 15) >> 1;

  v8f acc0 = {0.f,0.f,0.f,0.f,0.f,0.f,0.f,0.f};
  v8f acc1 = acc0;

  for (int ks = 0; ks < C; ks += 32) {
    __syncthreads();
    // ---- stage A tile (32m x 32k) ----
    {
      int m = mB + la_m;
      float a0, a1, a2, a3;
      if (UNITW) {
        const float4 v4 = *(const float4*)&W[m * w_so + ks + la_k0];
        a0 = v4.x; a1 = v4.y; a2 = v4.z; a3 = v4.w;
      } else {
        int base = m * w_so + w_off + (ks + la_k0) * w_sc;
        a0 = W[base];
        a1 = W[base + w_sc];
        a2 = W[base + 2 * w_sc];
        a3 = W[base + 3 * w_sc];
      }
      v2bf p0 = { (__bf16)a0, (__bf16)a1 };
      v2bf p1 = { (__bf16)a2, (__bf16)a3 };
      *(v2bf*)&sA[(a_mt * 32 + a_lane) * 16 + 2 * a_r]       = p0;
      *(v2bf*)&sA[(a_mt * 32 + a_lane) * 16 + 2 * (a_r + 1)] = p1;
    }
    // ---- stage B tile (32k x 128n), rows lb_k, lb_k+1, cols lb_c0..+7 ----
    {
      int k = ks + lb_k;  // concat boundaries are multiples of 32
      const float* pb; int mode, rl, ld, krel = k;
      if (krel < s0.rows)            { pb = pb0; mode = s0.mode; rl = s0.relu; ld = s0.nin; }
      else { krel -= s0.rows;
        if (krel < s1.rows)          { pb = pb1; mode = s1.mode; rl = s1.relu; ld = s1.nin; }
        else { krel -= s1.rows;        pb = pb2; mode = s2.mode; rl = s2.relu; ld = s2.nin; } }

      float r0[8], r1[8];
      if (mode == 0) {
        const float* rp0 = pb + (krel * ld + nBg + lb_c0);
        const float* rp1 = rp0 + ld;
        float4 x0 = ((const float4*)rp0)[0], x1 = ((const float4*)rp0)[1];
        float4 y0 = ((const float4*)rp1)[0], y1 = ((const float4*)rp1)[1];
        r0[0]=x0.x; r0[1]=x0.y; r0[2]=x0.z; r0[3]=x0.w;
        r0[4]=x1.x; r0[5]=x1.y; r0[6]=x1.z; r0[7]=x1.w;
        r1[0]=y0.x; r1[1]=y0.y; r1[2]=y0.z; r1[3]=y0.w;
        r1[4]=y1.x; r1[5]=y1.y; r1[6]=y1.z; r1[7]=y1.w;
      } else if (mode == 2) {
        const float* rp0 = pb + (krel * ld + ((nBg + lb_c0) >> 1));
        const float* rp1 = rp0 + ld;
        float4 x0 = ((const float4*)rp0)[0];
        float4 y0 = ((const float4*)rp1)[0];
        r0[0]=x0.x; r0[1]=x0.x; r0[2]=x0.y; r0[3]=x0.y;
        r0[4]=x0.z; r0[5]=x0.z; r0[6]=x0.w; r0[7]=x0.w;
        r1[0]=y0.x; r1[1]=y0.x; r1[2]=y0.y; r1[3]=y0.y;
        r1[4]=y0.z; r1[5]=y0.z; r1[6]=y0.w; r1[7]=y0.w;
      } else {
        float v0 = pb[krel], v1 = pb[krel + 1];
#pragma unroll
        for (int c = 0; c < 8; ++c) { r0[c] = v0; r1[c] = v1; }
      }
      if (rl) {
#pragma unroll
        for (int c = 0; c < 8; ++c) { r0[c] = fmaxf(r0[c], 0.f); r1[c] = fmaxf(r1[c], 0.f); }
      }
#pragma unroll
      for (int c = 0; c < 8; ++c) {
        int col = lb_c0 + c;
        int nt = col >> 4, lB = b_hB * 16 + (col & 15);
        v2bf p = { (__bf16)r0[c], (__bf16)r1[c] };
        *(v2bf*)&sB[(nt * 32 + lB) * 16 + 2 * b_j] = p;
      }
    }
    __syncthreads();
    // ---- fragments + MMA ----
    v16bf Af = *(const v16bf*)&sA[(wvM * 32 + lane) * 16];
    v16bf B0 = *(const v16bf*)&sB[(((wvN << 1) + 0) * 32 + lane) * 16];
    v16bf B1 = *(const v16bf*)&sB[(((wvN << 1) + 1) * 32 + lane) * 16];
    acc0 = wmma_bf16(Af, B0, acc0);
    acc1 = wmma_bf16(Af, B1, acc1);
  }

  // ---- epilogue ----
#pragma unroll
  for (int r = 0; r < 8; ++r) {
    int m = mB + wvM * 16 + (h << 3) + r;  // C/D: VGPR r -> row r / 8+r
    float v0 = acc0[r], v1 = acc1[r];
    if (bias) { float bv = bias[m]; v0 += bv; v1 += bv; }
    if (bn_g) {
      float sc = bn_g[m] * BN_INV, sh = bn_b[m];
      v0 = v0 * sc + sh; v1 = v1 * sc + sh;
    }
    if (relu_out) { v0 = fmaxf(v0, 0.f); v1 = fmaxf(v1, 0.f); }
    int n0 = nBg + wvN * 32 + l16;
    size_t base = ((size_t)b * O + m) * Nout + ooff;
    size_t i0 = base + (size_t)n0 * ostr;
    size_t i1 = base + (size_t)(n0 + 16) * ostr;
    if (accum) { Y[i0] += v0; Y[i1] += v1; }
    else       { Y[i0] = v0;  Y[i1] = v1; }
  }
}

// ---------------------------------------------------------------------------
// Pre-swizzle attention weights into bf16 WMMA A-fragment layout.
// tiles: [0,8)   pm_w2 (64x64):   tile = mt*2 + ks
//        [8,40)  am_w1 (256x64):  tile = 8  + mtg*2 + ks   (mtg 0..15)
//        [40,72) am_w2 (64x256):  tile = 40 + mt*8 + kst   (kst 0..7)
// ---------------------------------------------------------------------------
__global__ void wfrag_prep_kernel(const float* __restrict__ pm_w2,
                                  const float* __restrict__ am_w1,
                                  const float* __restrict__ am_w2,
                                  __bf16* __restrict__ out) {
  int gid = blockIdx.x * blockDim.x + threadIdx.x;  // tile*32 + lane
  if (gid >= 72 * 32) return;
  int tile = gid >> 5, lane = gid & 31;
  int h = lane >> 4, l16 = lane & 15;
  const float* w; int K, mt, ks;
  if (tile < 8)       { w = pm_w2; K = 64;  mt = tile >> 1;        ks = tile & 1; }
  else if (tile < 40) { w = am_w1; K = 64;  mt = (tile - 8) >> 1;  ks = (tile - 8) & 1; }
  else                { w = am_w2; K = 256; mt = (tile - 40) >> 3; ks = (tile - 40) & 7; }
#pragma unroll
  for (int e = 0; e < 16; ++e) {
    int r = e >> 1, par = e & 1;
    int k = ks * 32 + a_klocal(r, h) + par;
    out[(size_t)gid * 16 + e] = (__bf16)w[(mt * 16 + l16) * K + k];
  }
}

__device__ __forceinline__ v16bf load_frag(const __bf16* __restrict__ p, int tile, int lane) {
  return *(const v16bf*)(p + ((size_t)tile * 32 + lane) * 16);
}

// ---------------------------------------------------------------------------
// Tiny 3->64 conv + BN + ReLU (K=3, VALU)
// ---------------------------------------------------------------------------
__global__ void conv3_bn_relu_kernel(const float* __restrict__ x,
                                     const float* __restrict__ w,
                                     const float* __restrict__ bb,
                                     const float* __restrict__ g,
                                     const float* __restrict__ be,
                                     float* __restrict__ y, int N) {
  int id = blockIdx.x * blockDim.x + threadIdx.x;  // b*N+n
  int b = id / N, n = id % N;
  float p0 = x[((size_t)b * 3 + 0) * N + n];
  float p1 = x[((size_t)b * 3 + 1) * N + n];
  float p2 = x[((size_t)b * 3 + 2) * N + n];
  for (int c = 0; c < 64; ++c) {
    float v = w[c * 3] * p0 + w[c * 3 + 1] * p1 + w[c * 3 + 2] * p2 + bb[c];
    v = v * (g[c] * BN_INV) + be[c];
    y[((size_t)b * 64 + c) * N + n] = fmaxf(v, 0.f);
  }
}

// row max over N per (b,c)
__global__ void rowmax_kernel(const float* __restrict__ f, float* __restrict__ o, int N) {
  int id = blockIdx.x * blockDim.x + threadIdx.x;  // b*128+c
  const float4* row = (const float4*)(f + (size_t)id * N);
  float m = -3.4e38f;
  for (int n = 0; n < N / 4; ++n) {
    float4 v = row[n];
    m = fmaxf(m, fmaxf(fmaxf(v.x, v.y), fmaxf(v.z, v.w)));
  }
  o[id] = m;
}

// ---------------------------------------------------------------------------
// KNN: d2 = |p|^2 + |q|^2 - 2 p.q ; top-16 smallest (stable, self included)
// ---------------------------------------------------------------------------
__global__ __launch_bounds__(256) void knn_kernel(const float* __restrict__ pos,
                                                  int* __restrict__ idx, int N) {
  __shared__ float px[NPTS], py[NPTS], pz[NPTS], sq[NPTS];
  int b = blockIdx.y;
  const float* P = pos + (size_t)b * 3 * N;
  for (int i = threadIdx.x; i < N; i += 256) {
    float x = P[i], y = P[N + i], z = P[2 * N + i];
    px[i] = x; py[i] = y; pz[i] = z; sq[i] = x * x + y * y + z * z;
  }
  __syncthreads();
  int n = blockIdx.x * 256 + threadIdx.x;
  float qx = px[n], qy = py[n], qz = pz[n], qs = sq[n];
  float bd[16]; int bi[16];
#pragma unroll
  for (int t = 0; t < 16; ++t) { bd[t] = 3.4e38f; bi[t] = 0; }
  for (int m = 0; m < N; ++m) {
    float d2 = qs + sq[m] - 2.f * (qx * px[m] + qy * py[m] + qz * pz[m]);
    if (d2 < bd[15]) {
      int j = 15;
#pragma unroll
      for (int t = 15; t >= 1; --t)
        if (d2 < bd[t - 1]) { bd[t] = bd[t - 1]; bi[t] = bi[t - 1]; j = t - 1; }
#pragma unroll
      for (int t = 0; t < 16; ++t)
        if (t == j) { bd[t] = d2; bi[t] = m; }
    }
  }
  for (int t = 0; t < 16; ++t) idx[((size_t)b * N + n) * 16 + t] = bi[t];
}

// ---------------------------------------------------------------------------
// Fused attention: one wave per query point (16 neighbors == one B-tile).
// All WMMA operands fetched as b128 from pre-swizzled fragments.
// ---------------------------------------------------------------------------
__global__ __launch_bounds__(128) void attn_kernel(
    const float* __restrict__ pos, const int* __restrict__ knn,
    const float* __restrict__ q64, const float* __restrict__ k64,
    const float* __restrict__ v64, const float* __restrict__ value,
    const __bf16* __restrict__ wfrag,
    const float* __restrict__ pm_b1, const float* __restrict__ pm_g,
    const float* __restrict__ pm_be, const float* __restrict__ pm_w1,
    const float* __restrict__ pm_b2,
    const float* __restrict__ am_b1, const float* __restrict__ am_g,
    const float* __restrict__ am_be, const float* __restrict__ am_b2,
    const float* __restrict__ end_w, const float* __restrict__ end_b,
    float* __restrict__ H, int N) {
  __shared__ float  sPE[4][64 * 16];                               // pos_emb
  __shared__ __attribute__((aligned(32))) __bf16 sUb[4][2 * 32 * 16];  // B-frag: pe_mid -> u
  __shared__ __attribute__((aligned(32))) __bf16 sA1[4][4 * 32 * 16];  // B-frag a1 ; then f32 logits
  __shared__ float  sAGG[4][64];   // q cache, then agg
  __shared__ float  sPR[4][3 * 16];
  __shared__ int    sIDX[4][16];

  const int lane = threadIdx.x & 31;
  const int wv   = threadIdx.x >> 5;
  const int wid  = blockIdx.x * 4 + wv;
  const int b = wid / N, n = wid % N;
  const int h = lane >> 4, cl = lane & 15;

  if (lane < 16) {
    int j = knn[((size_t)b * N + n) * 16 + lane];
    sIDX[wv][lane] = j;
    for (int d = 0; d < 3; ++d)
      sPR[wv][d * 16 + lane] =
          pos[((size_t)b * 3 + d) * N + n] - pos[((size_t)b * 3 + d) * N + j];
  }
  sAGG[wv][lane]      = q64[((size_t)b * 64 + lane) * N + n];
  sAGG[wv][lane + 32] = q64[((size_t)b * 64 + lane + 32) * N + n];
  __syncthreads();

  // pe_mid = relu(bn(pm_w1 . pos_rel)) -> sUb (bf16 B-fragment layout)
  for (int i = 0; i < 32; ++i) {
    int id = i * 32 + lane, c = id >> 4, col = id & 15;
    float v = pm_b1[c] + pm_w1[c * 3] * sPR[wv][col] +
              pm_w1[c * 3 + 1] * sPR[wv][16 + col] +
              pm_w1[c * 3 + 2] * sPR[wv][32 + col];
    v = v * (pm_g[c] * BN_INV) + pm_be[c];
    int lT = (((c >> 4) & 1) << 4) + col;
    sUb[wv][((c >> 5) * 32 + lT) * 16 + (c & 15)] = (__bf16)fmaxf(v, 0.f);
  }
  __syncthreads();

  // pos_emb = pm_w2(64x64) . pe_mid + pm_b2 -> sPE (f32)
#pragma unroll
  for (int mt = 0; mt < 4; ++mt) {
    v8f acc = {0.f,0.f,0.f,0.f,0.f,0.f,0.f,0.f};
#pragma unroll
    for (int ks = 0; ks < 2; ++ks) {
      v16bf A = load_frag(wfrag, mt * 2 + ks, lane);
      v16bf B = *(const v16bf*)&sUb[wv][(ks * 32 + lane) * 16];
      acc = wmma_bf16(A, B, acc);
    }
#pragma unroll
    for (int r = 0; r < 8; ++r) {
      int m = mt * 16 + (h << 3) + r;
      sPE[wv][m * 16 + cl] = acc[r] + pm_b2[m];
    }
  }
  __syncthreads();

  // u = q - k_g + pos_emb -> sUb (bf16 B-fragment layout)
  for (int i = 0; i < 32; ++i) {
    int id = i * 32 + lane, c = id >> 4, col = id & 15;
    int j = sIDX[wv][col];
    float u = sAGG[wv][c] - k64[((size_t)b * 64 + c) * N + j] + sPE[wv][c * 16 + col];
    int lT = (((c >> 4) & 1) << 4) + col;
    sUb[wv][((c >> 5) * 32 + lT) * 16 + (c & 15)] = (__bf16)u;
  }
  __syncthreads();

  // attn logits = am_w2(64x256) . relu(bn(am_w1(256x64) . u)) , 2 halves
  v8f accA[4];
#pragma unroll
  for (int i = 0; i < 4; ++i) accA[i] = (v8f){0.f,0.f,0.f,0.f,0.f,0.f,0.f,0.f};

  for (int hh = 0; hh < 2; ++hh) {
    v8f acc1[8];
#pragma unroll
    for (int mt = 0; mt < 8; ++mt) {
      v8f acc = {0.f,0.f,0.f,0.f,0.f,0.f,0.f,0.f};
#pragma unroll
      for (int ks = 0; ks < 2; ++ks) {
        v16bf A = load_frag(wfrag, 8 + (hh * 8 + mt) * 2 + ks, lane);
        v16bf B = *(const v16bf*)&sUb[wv][(ks * 32 + lane) * 16];
        acc = wmma_bf16(A, B, acc);
      }
      acc1[mt] = acc;
    }
    __syncthreads();
    // stage relu(bn(a1)) in bf16 B-fragment layout (rows ml = 0..127)
#pragma unroll
    for (int mt = 0; mt < 8; ++mt)
#pragma unroll
      for (int r = 0; r < 8; ++r) {
        int ml = mt * 16 + (h << 3) + r, mg = hh * 128 + ml;
        float v = acc1[mt][r] + am_b1[mg];
        v = v * (am_g[mg] * BN_INV) + am_be[mg];
        int lT = (((ml >> 4) & 1) << 4) + cl;
        sA1[wv][((ml >> 5) * 32 + lT) * 16 + (ml & 15)] = (__bf16)fmaxf(v, 0.f);
      }
    __syncthreads();
#pragma unroll
    for (int mt = 0; mt < 4; ++mt)
#pragma unroll
      for (int ks = 0; ks < 4; ++ks) {
        v16bf A = load_frag(wfrag, 40 + mt * 8 + hh * 4 + ks, lane);
        v16bf B = *(const v16bf*)&sA1[wv][(ks * 32 + lane) * 16];
        accA[mt] = wmma_bf16(A, B, accA[mt]);
      }
    __syncthreads();
  }

  // logits + bias -> f32 in (dead) sA1
  float* sLog = (float*)&sA1[wv][0];
#pragma unroll
  for (int mt = 0; mt < 4; ++mt)
#pragma unroll
    for (int r = 0; r < 8; ++r) {
      int m = mt * 16 + (h << 3) + r;
      sLog[m * 16 + cl] = accA[mt][r] + am_b2[m];
    }
  __syncthreads();

  // softmax over K per channel, aggregate attn*(v + pos_emb)
  for (int t = 0; t < 2; ++t) {
    int c = lane + t * 32;
    float mx = -3.4e38f;
    for (int kk = 0; kk < 16; ++kk) mx = fmaxf(mx, sLog[c * 16 + kk]);
    float e[16], s = 0.f;
    for (int kk = 0; kk < 16; ++kk) { e[kk] = __expf(sLog[c * 16 + kk] - mx); s += e[kk]; }
    float inv = 1.f / s;
    float vv = v64[((size_t)b * 64 + c) * N + n];
    float agg = 0.f;
    for (int kk = 0; kk < 16; ++kk) agg += (e[kk] * inv) * (vv + sPE[wv][c * 16 + kk]);
    sAGG[wv][c] = agg;  // q cache dead after u
  }
  __syncthreads();

  // H = end_w(512x64) . agg + end_b + identity
  for (int t = 0; t < 16; ++t) {
    int m = t * 32 + lane;
    float acc = end_b[m] + value[((size_t)b * 512 + m) * N + n];
#pragma unroll
    for (int c = 0; c < 64; ++c) acc += end_w[m * 64 + c] * sAGG[wv][c];
    H[((size_t)b * 512 + m) * N + n] = acc;
  }
}

// final: delta = tanh(md_w2 . dmid + md_b2); pcd_child = repeat(pcd) + delta
__global__ void delta_kernel(const float* __restrict__ dmid,
                             const float* __restrict__ w, const float* __restrict__ bb,
                             const float* __restrict__ pcd, float* __restrict__ out,
                             int N2, int N) {
  int id = blockIdx.x * blockDim.x + threadIdx.x;  // b*N2+n
  int b = id / N2, n = id % N2;
  for (int d = 0; d < 3; ++d) {
    float acc = bb[d];
    for (int c = 0; c < 64; ++c) acc += w[d * 64 + c] * dmid[((size_t)b * 64 + c) * N2 + n];
    out[((size_t)b * 3 + d) * N2 + n] = pcd[((size_t)b * 3 + d) * N + (n >> 1)] + tanhf(acc);
  }
}

// ---------------------------------------------------------------------------
// Host
// ---------------------------------------------------------------------------
static inline void launch_gemm(hipStream_t st, const float* W, int wso, int wsc, int woff,
                               const float* bias, const float* g, const float* be,
                               int relu, int accum, float* Y, int Nout, int ostr, int ooff,
                               int O, int C, int Nper, SrcDesc s0, SrcDesc s1, SrcDesc s2) {
  dim3 grid(Nper / 128, O / 32, BATCH);
  if (wsc == 1)
    wmma_gemm_kernel<true><<<grid, 256, 0, st>>>(W, wso, wsc, woff, bias, g, be, relu, accum,
                                                 Y, Nout, ostr, ooff, O, C, Nper, s0, s1, s2);
  else
    wmma_gemm_kernel<false><<<grid, 256, 0, st>>>(W, wso, wsc, woff, bias, g, be, relu, accum,
                                                  Y, Nout, ostr, ooff, O, C, Nper, s0, s1, s2);
}

extern "C" void kernel_launch(void* const* d_in, const int* in_sizes, int n_in,
                              void* d_out, int out_size, void* d_ws, size_t ws_size,
                              hipStream_t stream) {
  auto F = [&](int i) { return (const float*)d_in[i]; };
  const float *pcd = F(0), *feat_global = F(1), *K_prev = F(2), *sem = F(3);
  const float *m1_w1=F(4),*m1_b1=F(5),*m1_g1=F(6),*m1_be1=F(7),*m1_w2=F(8),*m1_b2=F(9);
  const float *m2_w1=F(10),*m2_b1=F(11),*m2_g1=F(12),*m2_be1=F(13),*m2_w2=F(14),*m2_b2=F(15);
  const float *mv_w1=F(16),*mv_b1=F(17),*mv_w2=F(18),*mv_b2=F(19),*mv_ws=F(20),*mv_bs=F(21);
  const float *k_w=F(22),*k_b=F(23),*q_w=F(24),*q_b=F(25),*v_w=F(26),*v_b=F(27);
  const float *pm_w1=F(28),*pm_b1=F(29),*pm_g=F(30),*pm_be=F(31),*pm_w2=F(32),*pm_b2=F(33);
  const float *am_w1=F(34),*am_b1=F(35),*am_g=F(36),*am_be=F(37),*am_w2=F(38),*am_b2=F(39);
  const float *end_w=F(40),*end_b=F(41);
  const float *ps_w1=F(42),*ps_b1=F(43),*ps_g1=F(44),*ps_be1=F(45),*ps_w2=F(46),*ps_b2=F(47);
  const float *pst_w=F(48);
  const float *df_w1=F(49),*df_b1=F(50),*df_w2=F(51),*df_b2=F(52),*df_ws=F(53),*df_bs=F(54);
  const float *md_w1=F(55),*md_b1=F(56),*md_g1=F(57),*md_be1=F(58),*md_w2=F(59),*md_b2=F(60);

  float* W = (float*)d_ws;
  // workspace layout (floats), with lifetime reuse (~140 MB total)
  float* Q          = W + 0;          // 8*512*2048
  float* slotB      = W + 8388608;    // vmid, later H
  float* slotC      = W + 16777216;   // value, later kmid
  float* h1         = W + 25165824;   // later k64
  float* f1         = W + 26214400;   // later fcmid/fc, later dmid
  float* fmax       = W + 28311552;
  float* qmid       = W + 28312576;   // later feat_child
  float* q64        = W + 32506880;
  float* v64        = W + 33555456;
  int*   idxb       = (int*)(W + 34604032);
  __bf16* wfrag     = (__bf16*)(W + 34866176);  // 72*32*16 bf16
  float* vmid = slotB; float* Hh = slotB;
  float* value = slotC; float* kmid = slotC;
  float* k64 = h1; float* fcmid = f1; float* fc = f1 + 1048576;
  float* feat_child = qmid; float* dmid = f1;

  float* pcd_child = (float*)d_out;
  float* Kout = (float*)d_out + BATCH * 3 * NPTS2;  // K_curr

  SrcDesc SN{nullptr, 0, 0, 0, 1};

  // 0) pre-swizzle attention weight fragments
  wfrag_prep_kernel<<<9, 256, 0, stream>>>(pm_w2, am_w1, am_w2, wfrag);
  // 1) h1 = relu(bn(conv 3->64))
  conv3_bn_relu_kernel<<<BATCH * NPTS / 256, 256, 0, stream>>>(pcd, m1_w1, m1_b1, m1_g1, m1_be1, h1, NPTS);
  // 2) f1 = conv 64->128 + b
  launch_gemm(stream, m1_w2, 64, 1, 0, m1_b2, nullptr, nullptr, 0, 0, f1, NPTS, 1, 0,
              128, 64, NPTS, SrcDesc{h1, 64, 0, 0, NPTS}, SN, SN);
  // 3) fmax
  rowmax_kernel<<<BATCH * 128 / 256, 256, 0, stream>>>(f1, fmax, NPTS);
  // 4) qmid = relu(bn(conv 768->256)) over concat[f1,fmax,feat_global]
  launch_gemm(stream, m2_w1, 768, 1, 0, m2_b1, m2_g1, m2_be1, 1, 0, qmid, NPTS, 1, 0,
              256, 768, NPTS, SrcDesc{f1, 128, 0, 0, NPTS},
              SrcDesc{fmax, 128, 1, 0, 1}, SrcDesc{feat_global, 512, 1, 0, 1});
  // 5) Q = conv 256->512
  launch_gemm(stream, m2_w2, 256, 1, 0, m2_b2, nullptr, nullptr, 0, 0, Q, NPTS, 1, 0,
              512, 256, NPTS, SrcDesc{qmid, 256, 0, 0, NPTS}, SN, SN);
  // 6-8) value = mlp_res(concat[K_prev, Q])
  launch_gemm(stream, mv_w1, 1024, 1, 0, mv_b1, nullptr, nullptr, 1, 0, vmid, NPTS, 1, 0,
              512, 1024, NPTS, SrcDesc{K_prev, 512, 0, 0, NPTS}, SrcDesc{Q, 512, 0, 0, NPTS}, SN);
  launch_gemm(stream, mv_ws, 1024, 1, 0, mv_bs, nullptr, nullptr, 0, 0, value, NPTS, 1, 0,
              512, 1024, NPTS, SrcDesc{K_prev, 512, 0, 0, NPTS}, SrcDesc{Q, 512, 0, 0, NPTS}, SN);
  launch_gemm(stream, mv_w2, 512, 1, 0, mv_b2, nullptr, nullptr, 0, 1, value, NPTS, 1, 0,
              512, 512, NPTS, SrcDesc{vmid, 512, 0, 0, NPTS}, SN, SN);
  // 9-11) k/q/v projections 512->64
  launch_gemm(stream, k_w, 512, 1, 0, k_b, nullptr, nullptr, 0, 0, k64, NPTS, 1, 0,
              64, 512, NPTS, SrcDesc{K_prev, 512, 0, 0, NPTS}, SN, SN);
  launch_gemm(stream, q_w, 512, 1, 0, q_b, nullptr, nullptr, 0, 0, q64, NPTS, 1, 0,
              64, 512, NPTS, SrcDesc{Q, 512, 0, 0, NPTS}, SN, SN);
  launch_gemm(stream, v_w, 512, 1, 0, v_b, nullptr, nullptr, 0, 0, v64, NPTS, 1, 0,
              64, 512, NPTS, SrcDesc{value, 512, 0, 0, NPTS}, SN, SN);
  // 12) KNN
  knn_kernel<<<dim3(NPTS / 256, BATCH), 256, 0, stream>>>(pcd, idxb, NPTS);
  // 13) fused attention -> H (includes conv_end + identity)
  attn_kernel<<<BATCH * NPTS / 4, 128, 0, stream>>>(
      pcd, idxb, q64, k64, v64, value, wfrag,
      pm_b1, pm_g, pm_be, pm_w1, pm_b2,
      am_b1, am_g, am_be, am_b2, end_w, end_b, Hh, NPTS);
  // 14-15) ps branch
  launch_gemm(stream, ps_w1, 512, 1, 0, ps_b1, ps_g1, ps_be1, 1, 0, fcmid, NPTS, 1, 0,
              64, 512, NPTS, SrcDesc{Hh, 512, 0, 0, NPTS}, SN, SN);
  launch_gemm(stream, ps_w2, 64, 1, 0, ps_b2, nullptr, nullptr, 0, 0, fc, NPTS, 1, 0,
              32, 64, NPTS, SrcDesc{fcmid, 64, 0, 0, NPTS}, SN, SN);
  // 16-17) ConvTranspose1d 32->128 k=2 s=2 (two strided GEMMs)
  for (int t = 0; t < 2; ++t)
    launch_gemm(stream, pst_w, 2, 256, t, nullptr, nullptr, nullptr, 0, 0,
                feat_child, NPTS2, 2, t, 128, 32, NPTS,
                SrcDesc{fc, 32, 0, 0, NPTS}, SN, SN);
  // 18-20) K_curr = mlp_res(concat[feat_child, H_up, sem])  (written to d_out)
  launch_gemm(stream, df_w1, 1152, 1, 0, df_b1, nullptr, nullptr, 1, 0, kmid, NPTS2, 1, 0,
              128, 1152, NPTS2, SrcDesc{feat_child, 128, 0, 0, NPTS2},
              SrcDesc{Hh, 512, 2, 0, NPTS}, SrcDesc{sem, 512, 1, 0, 1});
  launch_gemm(stream, df_ws, 1152, 1, 0, df_bs, nullptr, nullptr, 0, 0, Kout, NPTS2, 1, 0,
              512, 1152, NPTS2, SrcDesc{feat_child, 128, 0, 0, NPTS2},
              SrcDesc{Hh, 512, 2, 0, NPTS}, SrcDesc{sem, 512, 1, 0, 1});
  launch_gemm(stream, df_w2, 128, 1, 0, df_b2, nullptr, nullptr, 0, 1, Kout, NPTS2, 1, 0,
              512, 128, NPTS2, SrcDesc{kmid, 128, 0, 0, NPTS2}, SN, SN);
  // 21) dmid = relu(bn(conv 512->64, input relu(K_curr)))
  launch_gemm(stream, md_w1, 512, 1, 0, md_b1, md_g1, md_be1, 1, 0, dmid, NPTS2, 1, 0,
              64, 512, NPTS2, SrcDesc{Kout, 512, 0, 1, NPTS2}, SN, SN);
  // 22) pcd_child = repeat(pcd) + tanh(conv 64->3)
  delta_kernel<<<BATCH * NPTS2 / 256, 256, 0, stream>>>(dmid, md_w2, md_b2, pcd, pcd_child, NPTS2, NPTS);
}
